// ConditionalStudentTeacherVGAE_42975442764640
// MI455X (gfx1250) — matrix operand order repaired and checked by reading, not compiled, not run on hardware
//
#include <hip/hip_runtime.h>
#include <hip/hip_bf16.h>

// ---------------------------------------------------------------------------
// MI455X (gfx1250) implementation of rebalance_label_homophily + exact top-K
// edge selection + 10 rounds of degree pruning.
//
// Bandwidth budget (23.3 TB/s HBM):
//   k_mass   : 1 read pass of adj (256MB) fused into a WMMA GEMM adj@onehot
//   k_scale  : read adj + adj^T tiles (512MB), write probs (256MB)
//   top-k    : 2 histogram passes over upper triangle (~256MB) + mark
//   10 iters : thresh (read adj 256MB + sparse probs) + prune (read adj/probs
//              tile pairs ~768MB, write adj 256MB)
//   total ~11 GB  =>  ~0.5 ms, memory bound everywhere. No host syncs;
//   all selection logic (homophily skip, thresholds) is computed on-device.
// ---------------------------------------------------------------------------

typedef __attribute__((ext_vector_type(16))) _Float16 v16h;
typedef __attribute__((ext_vector_type(8)))  float    v8f;

#define N_DIM 8192
#define C_DIM 16
#define TARGET_EDGES 131072u
#define MAX_DEG 64u
#define ROW_CAP 2048
#define TILE 32

// ---------------------------------------------------------------------------
// labels = argmax(label_probs, axis=1) (first-max tie break, like jnp.argmax)
// also emit the TRANSPOSED f16 one-hot matrix [class][node] so the WMMA B
// fragment is a single contiguous 32B vector load per lane per K-step.
// ---------------------------------------------------------------------------
__global__ void k_labels(const float* __restrict__ lp,
                         int* __restrict__ labels,
                         _Float16* __restrict__ onehotT) {
  int i = blockIdx.x * blockDim.x + threadIdx.x;
  if (i >= N_DIM) return;
  float best = lp[(size_t)i * C_DIM];
  int bi = 0;
  for (int c = 1; c < C_DIM; ++c) {
    float v = lp[(size_t)i * C_DIM + c];
    if (v > best) { best = v; bi = c; }
  }
  labels[i] = bi;
  for (int c = 0; c < C_DIM; ++c)
    onehotT[(size_t)c * N_DIM + i] = (c == bi) ? (_Float16)1.0f : (_Float16)0.0f;
}

// ---------------------------------------------------------------------------
// classsum = adj @ onehot via V_WMMA_F32_16X16X32_F16, one wave per 16 rows.
//   total_mass = sum(classsum)            (one-hot rows sum to 1)
//   same_mass  = sum_i classsum[i,labels[i]] - sum_i adj[i,i]
// Two independent accumulators over interleaved K streams so consecutive
// v_wmma have no C/D RAW and can issue back-to-back; also doubles MLP on the
// streaming A loads.
// A layout used here (16-bit A 16x32): lane l holds M=l&15,
//   K(e) = ((l>=16)?16:0) + e  -> 16 consecutive floats of the adj row.
// B layout uses the same contiguous K order as A: lane l holds
//   column N=l&15, K(e) = ((l>=16)?16:0) + e  -> contiguous in onehotT.
// ---------------------------------------------------------------------------
__device__ __forceinline__ v16h load_a_frag(const float* ap) {
  float4 f0 = *(const float4*)(ap);
  float4 f1 = *(const float4*)(ap + 4);
  float4 f2 = *(const float4*)(ap + 8);
  float4 f3 = *(const float4*)(ap + 12);
  v16h a;
  a[0]  = (_Float16)f0.x; a[1]  = (_Float16)f0.y; a[2]  = (_Float16)f0.z; a[3]  = (_Float16)f0.w;
  a[4]  = (_Float16)f1.x; a[5]  = (_Float16)f1.y; a[6]  = (_Float16)f1.z; a[7]  = (_Float16)f1.w;
  a[8]  = (_Float16)f2.x; a[9]  = (_Float16)f2.y; a[10] = (_Float16)f2.z; a[11] = (_Float16)f2.w;
  a[12] = (_Float16)f3.x; a[13] = (_Float16)f3.y; a[14] = (_Float16)f3.z; a[15] = (_Float16)f3.w;
  return a;
}

__global__ void k_mass(const float* __restrict__ adj,
                       const _Float16* __restrict__ onehotT,
                       const int* __restrict__ labels,
                       float* __restrict__ prm) {
  const int wave = threadIdx.x >> 5;
  const int lane = threadIdx.x & 31;
  const int rb   = blockIdx.x * 8 + wave;      // 512 row-blocks of 16 rows
  const int m    = lane & 15;
  const int kb   = (lane >> 4) * 16;           // A-matrix K base for this lane
  const int row  = rb * 16 + m;
  const int col  = lane & 15;                  // B-matrix class column
  const int kobB = (lane >> 4) * 16;           // B-matrix K base

  v8f acc0 = {}, acc1 = {};
  const size_t rbase = (size_t)row * N_DIM;
  const _Float16* bbase = onehotT + (size_t)col * N_DIM + kobB;

  for (int k0 = 0; k0 < N_DIM; k0 += 64) {
    const float* ap0 = adj + rbase + k0 + kb;
    const float* ap1 = ap0 + 32;
    // prefetch the streaming A operand a bit ahead (global_prefetch_b8)
    __builtin_prefetch((const void*)(ap0 + 2048), 0, 0);
    v16h a0 = load_a_frag(ap0);
    v16h a1 = load_a_frag(ap1);
    v16h b0 = *(const v16h*)(bbase + k0);        // 32B contiguous (2x b128)
    v16h b1 = *(const v16h*)(bbase + k0 + 32);
    acc0 = __builtin_amdgcn_wmma_f32_16x16x32_f16(
        false, a0, false, b0, (short)0, acc0, false, false);
    acc1 = __builtin_amdgcn_wmma_f32_16x16x32_f16(
        false, a1, false, b1, (short)0, acc1, false, false);
  }

  // C/D layout: lane l element v -> M = v + ((l>=16)?8:0), class col = l&15.
  float tot = 0.f, same = 0.f;
#pragma unroll
  for (int v = 0; v < 8; ++v) {
    int M = v + ((lane >> 4) * 8);
    float val = acc0[v] + acc1[v];
    tot += val;
    if (labels[rb * 16 + M] == col) same += val;
  }
  if (lane < 16) {                    // remove diagonal (same excludes i==j)
    int r2 = rb * 16 + lane;
    same -= adj[(size_t)r2 * N_DIM + r2];
  }
  for (int off = 16; off; off >>= 1) {
    tot  += __shfl_xor(tot,  off, 32);
    same += __shfl_xor(same, off, 32);
  }
  if (lane == 0) {
    atomicAdd(&prm[0], same);
    atomicAdd(&prm[1], tot);
  }
}

// ---------------------------------------------------------------------------
// Tiny scalar kernel: homophily weights, rescale factor, skip flag.
// mean(scaled) before clamp/diag-zero equals (ws*same + wd*diff)/N^2 because
// symmetrization preserves the total sum.
// ---------------------------------------------------------------------------
__global__ void k_params(float* prm) {
  float same  = prm[0];
  float total = prm[1];
  float diff  = total - same;
  float cur   = same / total;
  float ws = 0.7f / (cur + 1e-6f);
  ws = fminf(fmaxf(ws, 1.f / 3.f), 3.f);
  float wd = 0.3f / (1.f - cur + 1e-6f);
  wd = fminf(fmaxf(wd, 1.f / 3.f), 3.f);
  float invN2 = 1.f / ((float)N_DIM * (float)N_DIM);
  float meanAdj = total * invN2;
  float meanScaled = (ws * same + wd * diff) * invN2;
  float r = meanAdj / fmaxf(meanScaled, 1e-6f);
  int skip = (same <= 1e-6f) || (diff <= 1e-6f) || (fabsf(cur - 0.7f) <= 0.05f);
  prm[2] = ws; prm[3] = wd; prm[4] = r;
  ((int*)prm)[5] = skip;
}

// ---------------------------------------------------------------------------
// probs = rebalanced matrix. Symmetric tile-pair kernel: block (bi<=bj)
// stages both (bi,bj) and (bj,bi) 32x32 tiles in LDS so transposed reads and
// both writes are coalesced, and every element is touched exactly once.
// ---------------------------------------------------------------------------
__global__ void k_scale(const float* __restrict__ adj,
                        const int* __restrict__ labels,
                        const float* __restrict__ prm,
                        float* __restrict__ probs) {
  int bi = blockIdx.y, bj = blockIdx.x;
  if (bj < bi) return;
  __shared__ float la[TILE][TILE + 1], lb[TILE][TILE + 1];
  __shared__ int lli[TILE], llj[TILE];
  int tx = threadIdx.x & 31, r0 = threadIdx.x >> 5;
  for (int rr = r0; rr < TILE; rr += 8) {
    la[rr][tx] = adj[(size_t)(bi * TILE + rr) * N_DIM + bj * TILE + tx];
    lb[rr][tx] = adj[(size_t)(bj * TILE + rr) * N_DIM + bi * TILE + tx];
  }
  if (threadIdx.x < TILE) {
    lli[threadIdx.x] = labels[bi * TILE + threadIdx.x];
    llj[threadIdx.x] = labels[bj * TILE + threadIdx.x];
  }
  __syncthreads();
  float ws = prm[2], wd = prm[3], r = prm[4];
  int skip = ((const int*)prm)[5];
  for (int rr = r0; rr < TILE; rr += 8) {
    int i = bi * TILE + rr, j = bj * TILE + tx;
    float v;
    if (skip)        v = la[rr][tx];
    else if (i == j) v = 0.f;
    else {
      float w = (lli[rr] == llj[tx]) ? ws : wd;
      v = fmaxf(w * r * 0.5f * (la[rr][tx] + lb[tx][rr]), 0.f);
    }
    probs[(size_t)i * N_DIM + j] = v;
    if (bi != bj) {
      int j2 = bj * TILE + rr, i2 = bi * TILE + tx;
      float v2;
      if (skip) v2 = lb[rr][tx];
      else {
        float w2 = (llj[rr] == lli[tx]) ? ws : wd;
        v2 = fmaxf(w2 * r * 0.5f * (lb[rr][tx] + la[tx][rr]), 0.f);
      }
      probs[(size_t)j2 * N_DIM + i2] = v2;
    }
  }
}

// ---------------------------------------------------------------------------
// Exact global top-K via 2-level radix histogram on float bit patterns
// (all probs >= 0, so unsigned-bit order == float order).
// ---------------------------------------------------------------------------
__global__ void k_hist(const float* __restrict__ probs,
                       unsigned* __restrict__ hist,
                       const unsigned* __restrict__ prmU, int pass) {
  unsigned prefix = pass ? prmU[8] : 0u;
  const long long total = (long long)N_DIM * N_DIM;
  for (long long idx = blockIdx.x * (long long)blockDim.x + threadIdx.x;
       idx < total; idx += (long long)gridDim.x * blockDim.x) {
    int i = (int)(idx >> 13), j = (int)(idx & (N_DIM - 1));
    if (j <= i) continue;                       // strict upper triangle
    unsigned bits = __float_as_uint(probs[idx]);
    if (!pass)
      atomicAdd(&hist[bits >> 16], 1u);
    else if ((bits >> 16) == prefix)
      atomicAdd(&hist[bits & 0xFFFFu], 1u);
  }
}

__global__ void k_scan(const unsigned* __restrict__ hist,
                       unsigned* __restrict__ prmU, int pass) {
  __shared__ unsigned csum[256];
  int t = threadIdx.x;
  unsigned s = 0;
  for (int b = 0; b < 256; ++b) s += hist[t * 256 + b];
  csum[t] = s;
  __syncthreads();
  if (t == 0) {
    unsigned above = pass ? prmU[9] : 0u;
    int csel = -1;
    for (int c = 255; c >= 0; --c) {
      if (above + csum[c] >= TARGET_EDGES) { csel = c; break; }
      above += csum[c];
    }
    unsigned binsel = 0;
    if (csel >= 0) {
      for (int b = 255; b >= 0; --b) {
        unsigned h = hist[csel * 256 + b];
        if (above + h >= TARGET_EDGES) { binsel = (unsigned)(csel * 256 + b); break; }
        above += h;
      }
    }
    if (!pass) { prmU[8] = binsel; prmU[9] = above; }        // prefix, count>
    else       { prmU[10] = (prmU[8] << 16) | binsel;        // exact threshold
                 prmU[11] = TARGET_EDGES - above; }          // tie quota
  }
}

// Mark selected upper-triangle edges; adj = upper + upper^T.
__global__ void k_mark(const float* __restrict__ probs,
                       unsigned* __restrict__ prmU,
                       float* __restrict__ out) {
  const unsigned T = prmU[10];
  const unsigned need = prmU[11];
  const long long total = (long long)N_DIM * N_DIM;
  for (long long idx = blockIdx.x * (long long)blockDim.x + threadIdx.x;
       idx < total; idx += (long long)gridDim.x * blockDim.x) {
    int i = (int)(idx >> 13), j = (int)(idx & (N_DIM - 1));
    if (j <= i) continue;
    unsigned bits = __float_as_uint(probs[idx]);
    bool set = false;
    if (bits > T) set = true;
    else if (bits == T) {
      unsigned n = atomicAdd(&prmU[12], 1u);
      set = (n < need);
    }
    if (set) {
      out[idx] = 1.f;
      out[(size_t)j * N_DIM + i] = 1.f;
    }
  }
}

// ---------------------------------------------------------------------------
// Per-row pruning threshold: compact neighbor probs to LDS, then 31-step
// bisection over bit patterns for the 64th-largest value. effthresh = -inf
// when deg <= 64 (row fully kept, matching the (deg<=64) OR in the ref).
// ---------------------------------------------------------------------------
__global__ void k_thresh(const float* __restrict__ adj,
                         const float* __restrict__ probs,
                         float* __restrict__ effthresh) {
  const int row = blockIdx.x;
  __shared__ float list[ROW_CAP];
  __shared__ unsigned cnt;
  __shared__ int cshared;
  if (threadIdx.x == 0) cnt = 0;
  __syncthreads();
  const size_t rbase = (size_t)row * N_DIM;
  for (int j = threadIdx.x; j < N_DIM; j += blockDim.x) {
    if (adj[rbase + j] > 0.f) {
      unsigned p = atomicAdd(&cnt, 1u);
      if (p < ROW_CAP) list[p] = probs[rbase + j];
    }
  }
  __syncthreads();
  unsigned deg = cnt;
  if (deg <= MAX_DEG) {
    if (threadIdx.x == 0) effthresh[row] = __uint_as_float(0xff800000u); // -inf
    return;
  }
  unsigned m = deg < ROW_CAP ? deg : ROW_CAP;
  unsigned lo = 0u, hi = 0x7f800000u;          // values are finite, >= 0
  for (int it = 0; it < 31; ++it) {
    unsigned mid = lo + ((hi - lo) >> 1);
    if (threadIdx.x == 0) cshared = 0;
    __syncthreads();
    int c = 0;
    for (unsigned q = threadIdx.x; q < m; q += blockDim.x)
      if (__float_as_uint(list[q]) >= mid) ++c;
    atomicAdd(&cshared, c);
    __syncthreads();
    if ((unsigned)cshared >= MAX_DEG) lo = mid; else hi = mid;
    __syncthreads();
  }
  if (threadIdx.x == 0) effthresh[row] = __uint_as_float(lo);
}

// ---------------------------------------------------------------------------
// adj = adj * keep * keep^T, symmetric tile-pair kernel (in-place safe: each
// element is read and written only by the block owning its tile pair).
// keep[i,j] = probs[i,j] >= effthresh[i]  (effthresh=-inf encodes deg<=64).
// ---------------------------------------------------------------------------
__global__ void k_prune(const float* __restrict__ probs,
                        const float* __restrict__ effthresh,
                        float* __restrict__ out) {
  int bi = blockIdx.y, bj = blockIdx.x;
  if (bj < bi) return;
  __shared__ float A[TILE][TILE + 1], B[TILE][TILE + 1];
  __shared__ float P[TILE][TILE + 1], Q[TILE][TILE + 1];
  __shared__ float thi[TILE], thj[TILE];
  int tx = threadIdx.x & 31, r0 = threadIdx.x >> 5;
  for (int rr = r0; rr < TILE; rr += 8) {
    size_t a1 = (size_t)(bi * TILE + rr) * N_DIM + bj * TILE + tx;
    size_t a2 = (size_t)(bj * TILE + rr) * N_DIM + bi * TILE + tx;
    A[rr][tx] = out[a1];   P[rr][tx] = probs[a1];
    B[rr][tx] = out[a2];   Q[rr][tx] = probs[a2];
  }
  if (threadIdx.x < TILE) {
    thi[threadIdx.x] = effthresh[bi * TILE + threadIdx.x];
    thj[threadIdx.x] = effthresh[bj * TILE + threadIdx.x];
  }
  __syncthreads();
  for (int rr = r0; rr < TILE; rr += 8) {
    float kij = (P[rr][tx] >= thi[rr]) ? 1.f : 0.f;
    float kji = (Q[tx][rr] >= thj[tx]) ? 1.f : 0.f;
    out[(size_t)(bi * TILE + rr) * N_DIM + bj * TILE + tx] = A[rr][tx] * kij * kji;
    if (bi != bj) {
      float k1 = (Q[rr][tx] >= thj[rr]) ? 1.f : 0.f;
      float k2 = (P[tx][rr] >= thi[tx]) ? 1.f : 0.f;
      out[(size_t)(bj * TILE + rr) * N_DIM + bi * TILE + tx] = B[rr][tx] * k1 * k2;
    }
  }
}

// ---------------------------------------------------------------------------
extern "C" void kernel_launch(void* const* d_in, const int* in_sizes, int n_in,
                              void* d_out, int out_size, void* d_ws, size_t ws_size,
                              hipStream_t stream) {
  const float* adj_probs   = (const float*)d_in[0];
  const float* label_probs = (const float*)d_in[1];
  float* out = (float*)d_out;

  // workspace layout
  char* ws = (char*)d_ws;
  float*     probs   = (float*)ws;                 size_t off = (size_t)N_DIM * N_DIM * 4;
  float*     effth   = (float*)(ws + off);         off += (size_t)N_DIM * 4;
  int*       labels  = (int*)(ws + off);           off += (size_t)N_DIM * 4;
  _Float16*  onehotT = (_Float16*)(ws + off);      off += (size_t)N_DIM * C_DIM * 2;
  unsigned*  hist    = (unsigned*)(ws + off);      off += 65536u * 4;
  float*     prm     = (float*)(ws + off);
  unsigned*  prmU    = (unsigned*)prm;

  // zero accumulators / counters (fresh every call: deterministic)
  hipMemsetAsync(prm, 0, 256, stream);

  // --- rebalance -----------------------------------------------------------
  k_labels<<<N_DIM / 256, 256, 0, stream>>>(label_probs, labels, onehotT);
  k_mass  <<<N_DIM / (16 * 8), 256, 0, stream>>>(adj_probs, onehotT, labels, prm);
  k_params<<<1, 1, 0, stream>>>(prm);
  dim3 tg(N_DIM / TILE, N_DIM / TILE);
  k_scale <<<tg, 256, 0, stream>>>(adj_probs, labels, prm, probs);

  // --- exact top-K edge selection -----------------------------------------
  hipMemsetAsync(hist, 0, 65536u * 4, stream);
  k_hist<<<2048, 256, 0, stream>>>(probs, hist, prmU, 0);
  k_scan<<<1, 256, 0, stream>>>(hist, prmU, 0);
  hipMemsetAsync(hist, 0, 65536u * 4, stream);
  k_hist<<<2048, 256, 0, stream>>>(probs, hist, prmU, 1);
  k_scan<<<1, 256, 0, stream>>>(hist, prmU, 1);
  hipMemsetAsync(out, 0, (size_t)N_DIM * N_DIM * 4, stream);
  k_mark<<<2048, 256, 0, stream>>>(probs, prmU, out);

  // --- degree pruning iterations ------------------------------------------
  for (int it = 0; it < 10; ++it) {
    k_thresh<<<N_DIM, 256, 0, stream>>>(out, probs, effth);
    k_prune <<<tg, 256, 0, stream>>>(probs, effth, out);
  }
}
